// LIFSG_34187939676747
// MI455X (gfx1250) — matrix operand order repaired and checked by reading, not compiled, not run on hardware
//
#include <hip/hip_runtime.h>

typedef __attribute__((ext_vector_type(16))) _Float16 v16h;
typedef __attribute__((ext_vector_type(4)))  _Float16 v4h;
typedef __attribute__((ext_vector_type(8)))  float    v8f;

#define B_    32
#define T_    1000
#define NIN   1024
#define NOUT  512
#define NK    32            // 32 WMMA K-steps of 32 cover K=1024
#define WAVES 8
#define OWG   (WAVES * 16)  // 128 outputs per workgroup
#define NTHR  (WAVES * 32)  // 256 threads
#define DECAY 0.9512294245007140f

__launch_bounds__(NTHR, 1)
__global__ void lif_wmma_kernel(const float* __restrict__ in,
                                const float* __restrict__ W,
                                float* __restrict__ out)
{
    __shared__ v16h  sW[WAVES][NK][32];   // weight B-fragments, f16: 256 KB
    __shared__ v16h  sA[NK][32];          // spike A-fragments (16 timesteps): 32 KB
    __shared__ float sI[WAVES][16][16];   // accumulator spill for the scan: 8 KB
    __shared__ float sS[16][OWG];         // spike-output staging: 8 KB

    const int tid  = threadIdx.x;
    const int lane = tid & 31;
    const int wave = tid >> 5;
    const int b    = blockIdx.y;
    const int o_wg = blockIdx.x * OWG;

    // ---- stage W once: f32 -> f16, WMMA B layout, vectorized x4 ----
    // B (K=32 x N=16), contiguous k order used by this code: lanes 0-15 col N=lane hold K=0..15; lanes 16-31 K=16..31.
    for (int g = tid; g < OWG * NIN / 4; g += NTHR) {
        int e   = g << 2;
        int o_l = e >> 10;            // 0..127
        int i   = e & (NIN - 1);      // multiple of 4
        const float4 f = *(const float4*)(W + (size_t)(o_wg + o_l) * NIN + i);
        v4h pk = {(_Float16)f.x, (_Float16)f.y, (_Float16)f.z, (_Float16)f.w};
        int wv  = o_l >> 4, o16 = o_l & 15;
        int kk  = i >> 5,   ko  = i & 31;
        int ln  = o16 + ((ko >= 16) ? 16 : 0);
        int h   = ko & 15;            // multiple of 4 -> 8B-aligned slot
        *(v4h*)((_Float16*)&sW[wv][kk][ln] + h) = pk;
    }
    __syncthreads();

    const float* in_b = in + (size_t)b * T_ * NIN;
    float v = 0.0f;   // membrane potential carried across chunks (lanes 0..15)

    for (int t0 = 0; t0 < T_; t0 += 16) {
        // ---- stage 16 timesteps of spikes: f32 -> f16 A layout, vectorized x4 ----
        // A (M=16 x K=32), same contiguous k order as B: lanes 0-15: K=0..15 ; lanes 16-31: K=16..31
        for (int g = tid; g < 16 * NIN / 4; g += NTHR) {
            int e  = g << 2;
            int tl = e >> 10;
            int i  = e & (NIN - 1);
            int t  = t0 + tl;
            float4 f = make_float4(0.f, 0.f, 0.f, 0.f);
            if (t < T_) f = *(const float4*)(in_b + (size_t)t * NIN + i);
            v4h pk = {(_Float16)f.x, (_Float16)f.y, (_Float16)f.z, (_Float16)f.w};
            int kk = i >> 5, ko = i & 31;
            int ln = tl + ((ko >= 16) ? 16 : 0);
            int h  = ko & 15;   // multiple of 4
            *(v4h*)((_Float16*)&sA[kk][ln] + h) = pk;
        }
        // prefetch next chunk (256 threads x 64 floats = 16K floats)
        if (t0 + 16 < T_)
            __builtin_prefetch(in_b + (size_t)(t0 + 16) * NIN + tid * 64, 0, 1);
        __syncthreads();

        // ---- I tile (16t x 16o): two independent WMMA chains over K ----
        v8f c0 = {}, c1 = {};
        #pragma unroll 4
        for (int kk = 0; kk < NK; kk += 2) {
            v16h a0 = sA[kk][lane],     b0 = sW[wave][kk][lane];
            v16h a1 = sA[kk + 1][lane], b1 = sW[wave][kk + 1][lane];
            c0 = __builtin_amdgcn_wmma_f32_16x16x32_f16(false, a0, false, b0,
                                                        (short)0, c0, false, false);
            c1 = __builtin_amdgcn_wmma_f32_16x16x32_f16(false, a1, false, b1,
                                                        (short)0, c1, false, false);
        }
        v8f c = c0 + c1;

        // spill accumulator (lane n = col n; rows r / r+8) into (t, o) order
        #pragma unroll
        for (int r = 0; r < 8; ++r)
            sI[wave][r + ((lane >> 4) << 3)][lane & 15] = c[r];
        __syncthreads();

        // ---- serial LIF scan over the 16 timesteps (one lane per output) ----
        if (lane < 16) {
            #pragma unroll
            for (int tt = 0; tt < 16; ++tt) {
                if (t0 + tt < T_) {
                    v = v * DECAY + sI[wave][tt][lane];
                    float sp = (v > 1.0f) ? 1.0f : 0.0f;  // spike(v - V_THRESH)
                    v *= (1.0f - sp);                     // detached reset
                    sS[tt][wave * 16 + lane] = sp;
                }
            }
        }
        __syncthreads();

        // ---- coalesced store: rows of 128 consecutive floats (512 B) ----
        for (int idx = tid; idx < 16 * OWG; idx += NTHR) {
            int tt = idx >> 7, oo = idx & (OWG - 1);
            if (t0 + tt < T_)
                out[((size_t)b * T_ + t0 + tt) * NOUT + o_wg + oo] = sS[tt][oo];
        }
        __syncthreads();
    }
}

extern "C" void kernel_launch(void* const* d_in, const int* in_sizes, int n_in,
                              void* d_out, int out_size, void* d_ws, size_t ws_size,
                              hipStream_t stream) {
    (void)in_sizes; (void)n_in; (void)out_size; (void)d_ws; (void)ws_size;
    const float* in = (const float*)d_in[0];  // (B, T, n_in) float32 spikes
    const float* W  = (const float*)d_in[1];  // (n_out, n_in) float32
    float* out = (float*)d_out;               // (B, T, n_out) float32
    dim3 grid(NOUT / OWG, B_);                // (4, 32)
    lif_wmma_kernel<<<grid, NTHR, 0, stream>>>(in, W, out);
}
